// MyRobertaLMHead_63153199120848
// MI455X (gfx1250) — matrix operand order repaired and checked by reading, not compiled, not run on hardware
//
#include <hip/hip_runtime.h>
#include <hip/hip_bf16.h>
#include <math.h>

// ---------------------------------------------------------------------------
// MyRobertaLMHead fused kernel for gfx1250 (MI455X).
//   B=16, S=2048, D=1280, V=33
//   out[b,s,v] = (LN_ada(gelu(F @ Wd^T + bd)) ) @ Wout^T + out_bias
// bf16 WMMA (v_wmma_f32_16x16x32_bf16) for both GEMMs, f32 accumulation.
// GEMM1 runs two e-tiles per wave (shared A, two accumulator chains) for ILP.
// ---------------------------------------------------------------------------

typedef __bf16 bf16_t;
typedef __attribute__((ext_vector_type(16))) __bf16 bf16x16;
typedef __attribute__((ext_vector_type(8)))  __bf16 bf16x8;
typedef __attribute__((ext_vector_type(8)))  float  f32x8;
typedef __attribute__((ext_vector_type(4)))  float  f32x4;

#define D_DIM 1280
#define S_DIM 2048
#define B_DIM 16
#define V_DIM 33
#define VPAD  48          // V padded to 3 WMMA n-tiles
#define TS    16          // S-rows per block
#define FSTR  (D_DIM + 8) // bf16 LDS row stride (pad -> conflict-free b128)
#define XSTR  (D_DIM + 1) // f32 LDS row stride
#define KTILE 32
#define NWAVE 8

static __device__ __forceinline__ bf16x16 cat16(bf16x8 lo, bf16x8 hi) {
    return __builtin_shufflevector(lo, hi, 0,1,2,3,4,5,6,7,8,9,10,11,12,13,14,15);
}

// ---------------------------------------------------------------------------
// Kernel 1: convert dense_w -> bf16, W_out -> bf16 padded to 48 rows (zeros).
// ---------------------------------------------------------------------------
__global__ void cvt_weights_kernel(const float* __restrict__ dw,
                                   const float* __restrict__ wout,
                                   bf16_t* __restrict__ dwb,
                                   bf16_t* __restrict__ woutb) {
    int i = blockIdx.x * 256 + threadIdx.x;
    if (i < D_DIM * D_DIM) dwb[i] = (bf16_t)dw[i];
    if (i < VPAD * D_DIM) {
        int v = i / D_DIM;
        woutb[i] = (v < V_DIM) ? (bf16_t)wout[i] : (bf16_t)0.0f;
    }
}

// ---------------------------------------------------------------------------
// Kernel 2: AdaLN modulation (pre-folded with ln_weight / ln_bias).
//   gb[b,e] = sum_d silu(c[b,d]) * ada_w[e,d] + ada_b[e],  e in [0,2D)
//   gammaW[b,d] = ln_w[d] + gb[b,d];  betaB[b,d] = ln_b[d] + gb[b,D+d]
// ---------------------------------------------------------------------------
__global__ void adaln_kernel(const float* __restrict__ c,
                             const float* __restrict__ ada_w,
                             const float* __restrict__ ada_b,
                             const float* __restrict__ ln_w,
                             const float* __restrict__ ln_b,
                             float* __restrict__ gammaW,
                             float* __restrict__ betaB) {
    int widx = blockIdx.x * NWAVE + (threadIdx.x >> 5);
    int lane = threadIdx.x & 31;
    int b = widx / (2 * D_DIM);
    int e = widx % (2 * D_DIM);
    const float* crow = c + (size_t)b * D_DIM;
    const float* arow = ada_w + (size_t)e * D_DIM;
    float acc = 0.0f;
    #pragma unroll 4
    for (int i = lane; i < D_DIM; i += 32) {
        float cv = crow[i];
        float s  = cv / (1.0f + __expf(-cv));   // silu
        acc = fmaf(s, arow[i], acc);
    }
    #pragma unroll
    for (int off = 16; off > 0; off >>= 1) acc += __shfl_xor(acc, off, 32);
    if (lane == 0) {
        float v = acc + ada_b[e];
        if (e < D_DIM) gammaW[(size_t)b * D_DIM + e]           = ln_w[e] + v;
        else           betaB[(size_t)b * D_DIM + (e - D_DIM)]  = ln_b[e - D_DIM] + v;
    }
}

// ---------------------------------------------------------------------------
// Kernel 3: fused dense+GELU+AdaLN-LN+output projection, one 16-row tile/block.
// ---------------------------------------------------------------------------
__global__ void __launch_bounds__(256)
lmhead_main_kernel(const float* __restrict__ features,
                   const bf16_t* __restrict__ dwb,      // [D][D] bf16
                   const float* __restrict__ dense_b,   // [D]
                   const bf16_t* __restrict__ woutb,    // [VPAD][D] bf16
                   const float* __restrict__ out_bias,  // [V]
                   const float* __restrict__ gammaW,    // [B][D]
                   const float* __restrict__ betaB,     // [B][D]
                   float* __restrict__ out) {           // [B*S][V]
    __shared__ __align__(16) bf16_t ldsF[TS * FSTR];    // feature tile / normalized tile
    __shared__ __align__(16) float  ldsX[TS * XSTR];    // post-GELU activations (f32)

    const int tid  = threadIdx.x;
    const int w    = tid >> 5;
    const int lane = tid & 31;
    const int tile = blockIdx.x;                 // 0 .. B*S/TS-1
    const int b    = (tile * TS) / S_DIM;

    // ---- stage feature tile -> bf16 LDS (non-temporal: streamed once) -------
    {
        const f32x4* F4 = reinterpret_cast<const f32x4*>(features + (size_t)tile * TS * D_DIM);
        #pragma unroll 4
        for (int i = tid; i < TS * D_DIM / 4; i += 256) {
            int r = (i * 4) / D_DIM;
            int d = (i * 4) % D_DIM;
            f32x4 f = __builtin_nontemporal_load(&F4[i]);
            bf16_t* p = &ldsF[r * FSTR + d];
            p[0] = (bf16_t)f[0]; p[1] = (bf16_t)f[1];
            p[2] = (bf16_t)f[2]; p[3] = (bf16_t)f[3];
        }
    }
    __syncthreads();

    const int lrow  = lane & 15;
    const int ka    = (lane < 16) ? 0 : 16;   // A-matrix K sub-offset (contiguous k = 16h + s, matches B)
    const int kb    = (lane < 16) ? 0 : 16;   // B-matrix K sub-offset (contiguous k = 16h + s, matches A)
    const int mbase = (lane < 16) ? 0 : 8;    // C/D row base per lane half
    const bf16_t* Arow = &ldsF[lrow * FSTR];

    // ---- GEMM1: X = gelu(F @ Wd^T + bd) -------------------------------------
    // Wave w owns e-tile PAIRS: cols [32*ep, 32*ep+32), ep = w, w+8, ... (5 iters)
    // One shared A stream, two B streams, two independent WMMA chains.
    for (int ep = w; ep < D_DIM / 32; ep += NWAVE) {
        const int e0 = ep * 32;
        const bf16_t* Brow0 = dwb + (size_t)(e0 + lrow)      * D_DIM + kb;
        const bf16_t* Brow1 = dwb + (size_t)(e0 + 16 + lrow) * D_DIM + kb;
        __builtin_prefetch(Brow0, 0, 1);
        __builtin_prefetch(Brow1, 0, 1);
        f32x8 acc0 = {};
        f32x8 acc1 = {};
        #pragma unroll 2
        for (int kk = 0; kk < D_DIM / KTILE; ++kk) {
            const int d0 = kk * KTILE;
            bf16x8 a0 = *(const bf16x8*)(Arow + d0 + ka);
            bf16x8 a1 = *(const bf16x8*)(Arow + d0 + ka + 8);
            bf16x16 A = cat16(a0, a1);
            bf16x8 p0 = *(const bf16x8*)(Brow0 + d0);
            bf16x8 p1 = *(const bf16x8*)(Brow0 + d0 + 8);
            bf16x8 q0 = *(const bf16x8*)(Brow1 + d0);
            bf16x8 q1 = *(const bf16x8*)(Brow1 + d0 + 8);
            acc0 = __builtin_amdgcn_wmma_f32_16x16x32_bf16(
                false, A, false, cat16(p0, p1), (short)0, acc0, false, false);
            acc1 = __builtin_amdgcn_wmma_f32_16x16x32_bf16(
                false, A, false, cat16(q0, q1), (short)0, acc1, false, false);
        }
        const float bias0 = dense_b[e0 + lrow];
        const float bias1 = dense_b[e0 + 16 + lrow];
        #pragma unroll
        for (int j = 0; j < 8; ++j) {
            float x0 = acc0[j] + bias0;
            float x1 = acc1[j] + bias1;
            float g0 = 0.5f * x0 * (1.0f + erff(x0 * 0.70710678118654752f)); // exact gelu
            float g1 = 0.5f * x1 * (1.0f + erff(x1 * 0.70710678118654752f));
            ldsX[(mbase + j) * XSTR + (e0 + lrow)]      = g0;
            ldsX[(mbase + j) * XSTR + (e0 + 16 + lrow)] = g1;
        }
    }
    __syncthreads();   // all of ldsF consumed, all of ldsX produced

    // ---- LayerNorm + AdaLN affine; write normalized bf16 back into ldsF -----
    {
        const float* gW = gammaW + (size_t)b * D_DIM;
        const float* bB = betaB  + (size_t)b * D_DIM;
        #pragma unroll
        for (int rr = 0; rr < 2; ++rr) {
            const int r = 2 * w + rr;
            float s = 0.0f, s2 = 0.0f;
            for (int i = lane; i < D_DIM; i += 32) {
                float x = ldsX[r * XSTR + i];
                s += x; s2 = fmaf(x, x, s2);
            }
            #pragma unroll
            for (int off = 16; off > 0; off >>= 1) {
                s  += __shfl_xor(s,  off, 32);
                s2 += __shfl_xor(s2, off, 32);
            }
            const float mu  = s * (1.0f / D_DIM);
            const float var = s2 * (1.0f / D_DIM) - mu * mu;
            const float rs  = rsqrtf(var + 1e-5f);
            for (int i = lane; i < D_DIM; i += 32) {
                float x = ldsX[r * XSTR + i];
                float y = fmaf((x - mu) * rs, gW[i], bB[i]);
                ldsF[r * FSTR + i] = (bf16_t)y;
            }
        }
    }
    __syncthreads();   // normalized bf16 tile ready

    // ---- GEMM2: out = Xhat @ Wout^T + out_bias (3 n-tiles cover V=33) -------
    if (w < VPAD / 16) {                       // wave-uniform -> EXEC all-1s for WMMA
        const int v0 = w * 16;
        const bf16_t* Brow = woutb + (size_t)(v0 + lrow) * D_DIM + kb;
        f32x8 acc = {};
        #pragma unroll 4
        for (int kk = 0; kk < D_DIM / KTILE; ++kk) {
            const int d0 = kk * KTILE;
            bf16x8 a0 = *(const bf16x8*)(Arow + d0 + ka);
            bf16x8 a1 = *(const bf16x8*)(Arow + d0 + ka + 8);
            bf16x8 b0 = *(const bf16x8*)(Brow + d0);
            bf16x8 b1 = *(const bf16x8*)(Brow + d0 + 8);
            acc = __builtin_amdgcn_wmma_f32_16x16x32_bf16(
                false, cat16(a0, a1), false, cat16(b0, b1),
                (short)0, acc, false, false);
        }
        const int vcol = v0 + lrow;
        if (vcol < V_DIM) {
            const float ob = out_bias[vcol];
            #pragma unroll
            for (int j = 0; j < 8; ++j) {
                const int g = tile * TS + mbase + j;
                __builtin_nontemporal_store(acc[j] + ob, &out[(size_t)g * V_DIM + vcol]);
            }
        }
    }
}

// ---------------------------------------------------------------------------
extern "C" void kernel_launch(void* const* d_in, const int* in_sizes, int n_in,
                              void* d_out, int out_size, void* d_ws, size_t ws_size,
                              hipStream_t stream) {
    const float* features = (const float*)d_in[0];
    const float* c        = (const float*)d_in[1];
    const float* dense_w  = (const float*)d_in[2];
    const float* dense_b  = (const float*)d_in[3];
    const float* ln_w     = (const float*)d_in[4];
    const float* ln_b     = (const float*)d_in[5];
    const float* ada_w    = (const float*)d_in[6];
    const float* ada_b    = (const float*)d_in[7];
    const float* W_out    = (const float*)d_in[8];
    const float* out_bias = (const float*)d_in[9];
    float* out = (float*)d_out;

    // workspace layout (256B-aligned sections, ~3.6 MB total)
    char* ws = (char*)d_ws;
    bf16_t* dwb    = (bf16_t*)(ws);                                  // 1280*1280*2
    bf16_t* woutb  = (bf16_t*)(ws + 3276800);                        // 48*1280*2
    float*  gammaW = (float*) (ws + 3276800 + 122880);               // 16*1280*4
    float*  betaB  = (float*) (ws + 3276800 + 122880 + 81920);       // 16*1280*4

    // 1) weight conversion (covers max(1280*1280, 48*1280) elements)
    cvt_weights_kernel<<<(D_DIM * D_DIM + 255) / 256, 256, 0, stream>>>(
        dense_w, W_out, dwb, woutb);

    // 2) AdaLN gamma/beta (16 * 2560 wave-outputs, 8 waves per block)
    adaln_kernel<<<(B_DIM * 2 * D_DIM) / NWAVE, 256, 0, stream>>>(
        c, ada_w, ada_b, ln_w, ln_b, gammaW, betaB);

    // 3) fused main kernel: one 16-row tile per block
    lmhead_main_kernel<<<(B_DIM * S_DIM) / TS, 256, 0, stream>>>(
        features, dwb, dense_b, woutb, out_bias, gammaW, betaB, out);
}
